// PointTransformerBlock_42417097015912
// MI455X (gfx1250) — matrix operand order repaired and checked by reading, not compiled, not run on hardware
//
#include <hip/hip_runtime.h>
#include <hip/hip_bf16.h>
#include <math.h>

// ---------------- problem constants (match reference setup_inputs) ----------
#define NPTS 8192
#define NB   2
#define DIM  128
#define KNN  8
#define BN   (NB * NPTS)          // 16384 rows (b*N+n)
#define MOUT (NPTS / 4)           // 2048
#define D2   256
#define D4   512
#define ATTROWS  (BN * KNN)       // 131072
#define POOLROWS (NB * MOUT * KNN)// 32768
#define PKK  288                  // 259 padded up to multiple of 32

typedef _Float16 half_t;
typedef half_t v8h  __attribute__((ext_vector_type(8)));
typedef half_t v16h __attribute__((ext_vector_type(16)));
typedef float  v8f  __attribute__((ext_vector_type(8)));
typedef unsigned u32x4 __attribute__((ext_vector_type(4)));
typedef unsigned u32x8 __attribute__((ext_vector_type(8)));

__device__ __forceinline__ float gelu_exact(float v) {
    return 0.5f * v * (1.0f + erff(v * 0.70710678118f));
}

// ---------------------------------------------------------------------------
// Generic f16 WMMA GEMM: C[M,N] = A[M,K] * Bt[N,K]^T (+bias) (+x-residual)
// One wave -> 16x16 tile; block of 4 waves -> 16x64 tile.
// The 64xKd B tile is DMA'd to LDS once per block by the Tensor Data Mover
// (tensor_load_to_lds, tracked with TENSORcnt), then consumed via ds loads.
// A lane layout (16-bit A 16x32): lanes 0-15 row M=l, K in {0..15};
// lanes 16-31 row M=l-16, K in {16..31}  -> one 16-half chunk.
// B lane layout: lane L holds column N=L%16, 16 contiguous K at (L/16)*16.
// ---------------------------------------------------------------------------
__global__ void wmma_gemm_kernel(const half_t* __restrict__ A,
                                 const half_t* __restrict__ Bt,
                                 const float*  __restrict__ bias,
                                 float*        __restrict__ Cf,
                                 half_t*       __restrict__ Ch,
                                 const float*  __restrict__ addx, // (N,B,D) residual, or null
                                 int Mrows, int Ncols, int Kd, int act)
{
    extern __shared__ char lds_raw[];
    half_t* ldsB = (half_t*)lds_raw;           // 64 columns x Kd halfs

    const int lane = threadIdx.x & 31;
    const int wave = threadIdx.x >> 5;
    const int m0 = blockIdx.x << 4;
    const int n0blk = blockIdx.y << 6;

    // ---- TDM: DMA the 64xKd B tile (rows n0blk..n0blk+63 of Bt) into LDS ----
    if (threadIdx.x < 32) { // wave 0 only (uniform branch)
        const unsigned lds_lo = (unsigned)(size_t)ldsB; // low 32 bits = LDS offset
        const unsigned long long ga =
            (unsigned long long)(size_t)(Bt + (size_t)n0blk * Kd);
        u32x4 g0;
        g0[0] = 1u;                                   // count=1, user descriptor
        g0[1] = lds_lo;                               // lds_addr
        g0[2] = (unsigned)ga;                         // global_addr[31:0]
        g0[3] = (unsigned)(ga >> 32) | 0x80000000u;   // global_addr[56:32] | type=2
        u32x8 g1;
        g1[0] = 0x00010000u;                          // data_size=1 (2 bytes)
        g1[1] = ((unsigned)Kd & 0xFFFFu) << 16;       // tensor_dim0[15:0]
        g1[2] = (((unsigned)Kd >> 16) & 0xFFFFu)      // tensor_dim0[31:16]
              | (((unsigned)Ncols & 0xFFFFu) << 16);  // tensor_dim1[15:0]
        g1[3] = (((unsigned)Ncols >> 16) & 0xFFFFu)   // tensor_dim1[31:16]
              | (((unsigned)Kd & 0xFFFFu) << 16);     // tile_dim0 = Kd
        g1[4] = 64u;                                  // tile_dim1 = 64 rows (tile_dim2=0)
        g1[5] = (unsigned)Kd;                         // tensor_dim0_stride[31:0]
        g1[6] = 0u;                                   // stride0[47:32], stride1[15:0]
        g1[7] = 0u;
        asm volatile("tensor_load_to_lds %0, %1" :: "s"(g0), "s"(g1) : "memory");
        __builtin_amdgcn_s_wait_tensorcnt(0);
    }
    __syncthreads();

    const int ar   = m0 + (lane & 15);
    const int aoff = (lane >> 4) << 4;   // 0 or 16
    const int boff = (lane >> 4) << 4;   // 0 or 16
    const half_t* ap = A + (size_t)ar * Kd + aoff;
    const half_t* bp = ldsB + (size_t)((wave << 4) + (lane & 15)) * Kd + boff;

    v8f acc = {};
    for (int kk = 0; kk < Kd; kk += 32) {
        v16h a = *(const v16h*)(ap + kk);
        v16h b = *(const v16h*)(bp + kk);
        acc = __builtin_amdgcn_wmma_f32_16x16x32_f16(
                  false, a, false, b, (short)0, acc, false, false);
    }

    const int col   = n0blk + (wave << 4) + (lane & 15);
    const int rbase = m0 + ((lane >> 4) << 3);
    const float bv = bias ? bias[col] : 0.0f;
#pragma unroll
    for (int j = 0; j < 8; ++j) {
        const int row = rbase + j;
        float v = acc[j] + bv;
        if (addx) { // residual add from x laid out (N,B,D); row = b*N+n
            const int n  = row & (NPTS - 1);
            const int b_ = row >> 13;
            v += addx[((size_t)n * NB + b_) * DIM + col];
        }
        if (act == 1)      v = gelu_exact(v);
        else if (act == 2) v = (v >= 0.0f) ? v : 0.01f * v;
        if (Cf) Cf[(size_t)row * Ncols + col] = v;
        if (Ch) Ch[(size_t)row * Ncols + col] = (half_t)v;
    }
}

// ---------------- packing / transposition ----------------------------------
__global__ void pack_x_kernel(const float* __restrict__ x, half_t* __restrict__ xt16)
{
    int tid = blockIdx.x * blockDim.x + threadIdx.x;   // BN*DIM
    if (tid >= BN * DIM) return;
    int d = tid & (DIM - 1);
    int r = tid >> 7;                // b*N+n
    int n = r & (NPTS - 1);
    int b = r >> 13;
    xt16[tid] = (half_t)x[((size_t)n * NB + b) * DIM + d];
}

__global__ void transpose_w_kernel(const float* __restrict__ W,
                                   half_t* __restrict__ Wt, int Kd, int Nc)
{
    int tid = blockIdx.x * blockDim.x + threadIdx.x;
    if (tid >= Kd * Nc) return;
    int k = tid % Kd;
    int n = tid / Kd;
    Wt[(size_t)n * Kd + k] = (half_t)W[(size_t)k * Nc + n];
}

__global__ void transpose_wpool_kernel(const float* __restrict__ W,
                                       half_t* __restrict__ Wt)
{
    int tid = blockIdx.x * blockDim.x + threadIdx.x;   // 256*288
    if (tid >= D2 * PKK) return;
    int k = tid % PKK;
    int n = tid / PKK;
    Wt[(size_t)n * PKK + k] = (k < 259) ? (half_t)W[(size_t)k * D2 + n] : (half_t)0.0f;
}

// ---------------- KNN (top-8 by squared distance, ascending) ---------------
// Point tiles are staged into LDS with async global->LDS copies (ASYNCcnt).
__global__ void knn_kernel(const float* __restrict__ qpos,
                           const float* __restrict__ ppos,
                           int* __restrict__ outidx, int Mq)
{
    __shared__ float sp[256 * 3];
    const int b = blockIdx.y;
    const int q = blockIdx.x * blockDim.x + threadIdx.x;
    const bool valid = q < Mq;
    float qx = 0.f, qy = 0.f, qz = 0.f;
    if (valid) {
        const float* qp = qpos + ((size_t)b * Mq + q) * 3;
        qx = qp[0]; qy = qp[1]; qz = qp[2];
    }
    float bd[KNN]; int bi[KNN];
#pragma unroll
    for (int i = 0; i < KNN; ++i) { bd[i] = 3.4e38f; bi[i] = 0; }

    const unsigned lds_base = (unsigned)(size_t)&sp[0];
    for (int tile = 0; tile < NPTS; tile += 256) {
        __syncthreads();
        // async copy of 256 points (3072 bytes) straight into LDS: 192 x b128
        if (threadIdx.x < 192) {
            const unsigned lds_off = lds_base + threadIdx.x * 16u;
            const float* g = ppos + ((size_t)b * NPTS + tile) * 3 + threadIdx.x * 4;
            asm volatile("global_load_async_to_lds_b128 %0, %1, off"
                         :: "v"(lds_off), "v"(g) : "memory");
        }
        if (tile + 256 < NPTS)
            __builtin_prefetch(ppos + ((size_t)b * NPTS + tile + 256) * 3 + threadIdx.x, 0, 1);
        asm volatile("s_wait_asynccnt 0x0" ::: "memory");
        __syncthreads();
        if (valid) {
            for (int j = 0; j < 256; ++j) {
                float dx = qx - sp[j * 3 + 0];
                float dy = qy - sp[j * 3 + 1];
                float dz = qz - sp[j * 3 + 2];
                float d = dx * dx + dy * dy + dz * dz;
                if (d < bd[KNN - 1]) {
                    int pidx = tile + j;
                    int pos = KNN - 1;
                    while (pos > 0 && bd[pos - 1] > d) {
                        bd[pos] = bd[pos - 1]; bi[pos] = bi[pos - 1]; --pos;
                    }
                    bd[pos] = d; bi[pos] = pidx;
                }
            }
        }
    }
    if (valid) {
        int* o = outidx + ((size_t)b * Mq + q) * KNN;
#pragma unroll
        for (int i = 0; i < KNN; ++i) o[i] = bi[i];
    }
}

// ---------------- positional encoding hidden: gelu(dpos@Wpe1+b) ------------
__global__ void pe_hidden_kernel(const float* __restrict__ ppos,
                                 const int* __restrict__ idx1,
                                 const float* __restrict__ Wpe1,
                                 const float* __restrict__ bpe1,
                                 half_t* __restrict__ peh)
{
    const int row = blockIdx.x;          // (b*N+m)*KNN + kn
    const int c = threadIdx.x;           // 0..127
    const int bm = row >> 3;
    const int b = bm >> 13;
    const int m = bm & (NPTS - 1);
    const int j = idx1[row];
    const float* pm = ppos + ((size_t)b * NPTS + m) * 3;
    const float* pj = ppos + ((size_t)b * NPTS + j) * 3;
    const float dx = pm[0] - pj[0], dy = pm[1] - pj[1], dz = pm[2] - pj[2];
    float t = dx * Wpe1[c] + dy * Wpe1[DIM + c] + dz * Wpe1[2 * DIM + c] + bpe1[c];
    peh[(size_t)row * DIM + c] = (half_t)gelu_exact(t);
}

// ---------------- attention input: q - k_knn + pe --------------------------
__global__ void attn_in_kernel(const float* __restrict__ Q,
                               const float* __restrict__ Kf,
                               const float* __restrict__ pe,
                               const int* __restrict__ idx1,
                               half_t* __restrict__ ain)
{
    const int row = blockIdx.x;
    const int c = threadIdx.x;
    const int bm = row >> 3;
    const int b = bm >> 13;
    const int j = idx1[row];
    float v = Q[(size_t)bm * DIM + c]
            - Kf[((size_t)b * NPTS + j) * DIM + c]
            + pe[(size_t)row * DIM + c];
    ain[(size_t)row * DIM + c] = (half_t)v;
}

// ---------------- softmax over feature dim (128), with 1/sqrt(D) -----------
__global__ void softmax_kernel(float* __restrict__ attn)
{
    __shared__ float red[DIM];
    const int row = blockIdx.x;
    const int c = threadIdx.x;
    const float inv = 0.0883883476483184f; // 1/sqrt(128)
    float v = attn[(size_t)row * DIM + c] * inv;
    red[c] = v; __syncthreads();
    for (int s = 64; s > 0; s >>= 1) {
        if (c < s) red[c] = fmaxf(red[c], red[c + s]);
        __syncthreads();
    }
    const float mx = red[0]; __syncthreads();
    const float e = expf(v - mx);
    red[c] = e; __syncthreads();
    for (int s = 64; s > 0; s >>= 1) {
        if (c < s) red[c] += red[c + s];
        __syncthreads();
    }
    attn[(size_t)row * DIM + c] = e / red[0];
}

// ---------------- res = sum_k attn * (v_knn + pe) --------------------------
__global__ void res_kernel(const float* __restrict__ attn,
                           const float* __restrict__ Vf,
                           const float* __restrict__ pe,
                           const int* __restrict__ idx1,
                           half_t* __restrict__ res16)
{
    __shared__ int sj[KNN];
    const int bm = blockIdx.x;
    const int c = threadIdx.x;
    if (c < KNN) sj[c] = idx1[bm * KNN + c];
    __syncthreads();
    const int b = bm >> 13;
    float s = 0.f;
#pragma unroll
    for (int kn = 0; kn < KNN; ++kn) {
        const size_t row = (size_t)bm * KNN + kn;
        const float vv = Vf[((size_t)b * NPTS + sj[kn]) * DIM + c] + pe[row * DIM + c];
        s += attn[row * DIM + c] * vv;
    }
    res16[(size_t)bm * DIM + c] = (half_t)s;
}

// ---------------- RMSNorm (writes f16 GEMM operand) ------------------------
__global__ void rmsnorm_kernel(const float* __restrict__ x1,
                               const float* __restrict__ scale,
                               half_t* __restrict__ xn)
{
    __shared__ float red[DIM];
    const int r = blockIdx.x;
    const int c = threadIdx.x;
    const float v = x1[(size_t)r * DIM + c];
    red[c] = v * v; __syncthreads();
    for (int s = 64; s > 0; s >>= 1) {
        if (c < s) red[c] += red[c + s];
        __syncthreads();
    }
    const float rms = sqrtf(red[0]) * 0.0883883476483184f; // sqrt(sum)/sqrt(D)
    xn[(size_t)r * DIM + c] = (half_t)(scale[c] * (v / (rms + 1e-8f)));
}

// ---------------- farthest point sampling (sequential argmax) --------------
__global__ void fps_kernel(const float* __restrict__ ppos, float* __restrict__ fp)
{
    __shared__ float sval[1024];
    __shared__ int   sidx[1024];
    __shared__ int   slast;
    const int b = blockIdx.x;
    const int t = threadIdx.x;
    const float* P = ppos + (size_t)b * NPTS * 3;
    float mind[8];
#pragma unroll
    for (int u = 0; u < 8; ++u) mind[u] = 3.4e38f;
    int last = 0;
    if (t == 0) {
        float* o = fp + (size_t)b * MOUT * 3;
        o[0] = P[0]; o[1] = P[1]; o[2] = P[2];
    }
    for (int i = 1; i < MOUT; ++i) {
        const float lx = P[last * 3 + 0], ly = P[last * 3 + 1], lz = P[last * 3 + 2];
        float bestv = -1.0f; int besti = 0;
#pragma unroll
        for (int u = 0; u < 8; ++u) {
            const int j = t + u * 1024;
            const float dx = P[j * 3 + 0] - lx;
            const float dy = P[j * 3 + 1] - ly;
            const float dz = P[j * 3 + 2] - lz;
            const float d = dx * dx + dy * dy + dz * dz;
            if (d < mind[u]) mind[u] = d;
            if (mind[u] > bestv) { bestv = mind[u]; besti = j; }
        }
        sval[t] = bestv; sidx[t] = besti;
        __syncthreads();
        for (int s = 512; s > 0; s >>= 1) {
            if (t < s) {
                if (sval[t + s] > sval[t] ||
                    (sval[t + s] == sval[t] && sidx[t + s] < sidx[t])) {
                    sval[t] = sval[t + s]; sidx[t] = sidx[t + s];
                }
            }
            __syncthreads();
        }
        if (t == 0) {
            slast = sidx[0];
            float* o = fp + ((size_t)b * MOUT + i) * 3;
            o[0] = P[slast * 3 + 0]; o[1] = P[slast * 3 + 1]; o[2] = P[slast * 3 + 2];
        }
        __syncthreads();
        last = slast;
        __syncthreads();
    }
}

// ---------------- pool operand packing [fp(3) | knn_x(256) | pad(29)] ------
__global__ void pool_pack_kernel(const float* __restrict__ fp,
                                 const float* __restrict__ x2,
                                 const int* __restrict__ kidx2,
                                 half_t* __restrict__ Apool)
{
    const size_t tid = (size_t)blockIdx.x * blockDim.x + threadIdx.x;
    if (tid >= (size_t)POOLROWS * PKK) return;
    const int c = (int)(tid % PKK);
    const int prow = (int)(tid / PKK);   // (b*MOUT+m)*KNN+kn
    const int b = prow >> 14;            // MOUT*KNN = 16384
    const int mk = prow & 16383;
    const int m = mk >> 3;
    float v;
    if (c < 3) {
        v = fp[((size_t)b * MOUT + m) * 3 + c];
    } else if (c < 259) {
        const int j = kidx2[prow];
        v = x2[((size_t)b * NPTS + j) * D2 + (c - 3)];
    } else {
        v = 0.0f;
    }
    Apool[tid] = (half_t)v;
}

// ---------------- max over K neighbors, write (n_out, B, 2D) ---------------
__global__ void pool_max_kernel(const float* __restrict__ H, float* __restrict__ out)
{
    const int tid = blockIdx.x * blockDim.x + threadIdx.x; // NB*MOUT*D2
    if (tid >= NB * MOUT * D2) return;
    const int c = tid & (D2 - 1);
    const int bm = tid >> 8;            // b*MOUT+m
    const int b = bm >> 11;
    const int m = bm & (MOUT - 1);
    float mx = -3.4e38f;
#pragma unroll
    for (int kn = 0; kn < KNN; ++kn)
        mx = fmaxf(mx, H[((size_t)bm * KNN + kn) * D2 + c]);
    out[((size_t)m * NB + b) * D2 + c] = mx;
}

// ---------------------------------------------------------------------------
extern "C" void kernel_launch(void* const* d_in, const int* in_sizes, int n_in,
                              void* d_out, int out_size, void* d_ws, size_t ws_size,
                              hipStream_t stream)
{
    const float* x     = (const float*)d_in[0];
    const float* p_pos = (const float*)d_in[1];
    const float* Wq = (const float*)d_in[2];  const float* bq = (const float*)d_in[3];
    const float* Wk = (const float*)d_in[4];  const float* bk = (const float*)d_in[5];
    const float* Wv = (const float*)d_in[6];  const float* bv = (const float*)d_in[7];
    const float* Wpe1 = (const float*)d_in[8];  const float* bpe1 = (const float*)d_in[9];
    const float* Wpe2 = (const float*)d_in[10]; const float* bpe2 = (const float*)d_in[11];
    const float* Wg = (const float*)d_in[12]; const float* bg = (const float*)d_in[13];
    const float* Wo = (const float*)d_in[14]; const float* bo = (const float*)d_in[15];
    const float* scale = (const float*)d_in[16];
    const float* Wf1 = (const float*)d_in[17]; const float* bf1 = (const float*)d_in[18];
    const float* Wf2 = (const float*)d_in[19]; const float* bf2 = (const float*)d_in[20];
    const float* Wpool = (const float*)d_in[21];

    char* ws = (char*)d_ws;
    // -------- workspace layout (all offsets 256B aligned) --------
    const size_t OFF_XT16   = 0;                       // BN*DIM f16     = 4  MiB
    const size_t OFF_WTQ    = 4194304;                 // 128x128 f16
    const size_t OFF_WTK    = OFF_WTQ + 32768;
    const size_t OFF_WTV    = OFF_WTK + 32768;
    const size_t OFF_WTPE2  = OFF_WTV + 32768;
    const size_t OFF_WTG    = OFF_WTPE2 + 32768;
    const size_t OFF_WTO    = OFF_WTG + 32768;
    const size_t OFF_WTF1   = OFF_WTO + 32768;         // 512x128 f16
    const size_t OFF_WTF2   = OFF_WTF1 + 131072;       // 256x512 f16
    const size_t OFF_WTPOOL = OFF_WTF2 + 262144;       // 256x288 f16
    const size_t OFF_Q      = 4931584;                 // BN*DIM f32 = 8 MiB
    const size_t OFF_K      = OFF_Q + 8388608;
    const size_t OFF_V      = OFF_K + 8388608;
    const size_t OFF_IDX1   = OFF_V + 8388608;         // BN*KNN int
    const size_t OFF_PEH16  = OFF_IDX1 + 524288;       // ATTROWS*DIM f16 (reused: attn_in16)
    const size_t OFF_PE     = OFF_PEH16 + 33554432;    // ATTROWS*DIM f32 (reused: Hpool)
    const size_t OFF_ATTN   = OFF_PE + 67108864;       // ATTROWS*DIM f32 (reused: Apool16)
    const size_t OFF_RES16  = OFF_ATTN + 67108864;     // BN*DIM f16
    const size_t OFF_X1     = OFF_RES16 + 4194304;     // BN*DIM f32
    const size_t OFF_XN16   = OFF_X1 + 8388608;        // BN*DIM f16
    const size_t OFF_F116   = OFF_XN16 + 4194304;      // BN*D4 f16
    const size_t OFF_X2     = OFF_F116 + 16777216;     // BN*D2 f32
    const size_t OFF_KIDX2  = OFF_X2 + 16777216;       // POOLROWS int

    half_t* xt16   = (half_t*)(ws + OFF_XT16);
    half_t* wtq    = (half_t*)(ws + OFF_WTQ);
    half_t* wtk    = (half_t*)(ws + OFF_WTK);
    half_t* wtv    = (half_t*)(ws + OFF_WTV);
    half_t* wtpe2  = (half_t*)(ws + OFF_WTPE2);
    half_t* wtg    = (half_t*)(ws + OFF_WTG);
    half_t* wto    = (half_t*)(ws + OFF_WTO);
    half_t* wtf1   = (half_t*)(ws + OFF_WTF1);
    half_t* wtf2   = (half_t*)(ws + OFF_WTF2);
    half_t* wtpool = (half_t*)(ws + OFF_WTPOOL);
    float*  Qf     = (float*)(ws + OFF_Q);
    float*  Kf     = (float*)(ws + OFF_K);
    float*  Vf     = (float*)(ws + OFF_V);
    int*    idx1   = (int*)(ws + OFF_IDX1);
    half_t* peh16  = (half_t*)(ws + OFF_PEH16);
    half_t* ain16  = (half_t*)(ws + OFF_PEH16);   // reuse after pe GEMM
    float*  pe_f   = (float*)(ws + OFF_PE);
    float*  attn_f = (float*)(ws + OFF_ATTN);
    half_t* res16  = (half_t*)(ws + OFF_RES16);
    float*  x1     = (float*)(ws + OFF_X1);
    half_t* xn16   = (half_t*)(ws + OFF_XN16);
    half_t* f116   = (half_t*)(ws + OFF_F116);
    float*  x2f    = (float*)(ws + OFF_X2);
    int*    kidx2  = (int*)(ws + OFF_KIDX2);
    half_t* apool  = (half_t*)(ws + OFF_ATTN);    // reuse after res
    float*  hpool  = (float*)(ws + OFF_PE);       // reuse after res

    float* out_h  = (float*)d_out;                  // (MOUT, B, D2)
    float* out_fp = out_h + (size_t)MOUT * NB * D2; // (B, MOUT, 3)

    const dim3 B128(128);
    auto cdiv = [](size_t a, size_t b) { return (unsigned)((a + b - 1) / b); };
    auto blds = [](int Kd) { return (size_t)(64 * Kd * 2); }; // dynamic LDS for B tile

    // 1) pack activations + weights to f16 (weights transposed to (N,K))
    pack_x_kernel<<<cdiv((size_t)BN * DIM, 256), 256, 0, stream>>>(x, xt16);
    transpose_w_kernel<<<cdiv(DIM * DIM, 256), 256, 0, stream>>>(Wq, wtq, DIM, DIM);
    transpose_w_kernel<<<cdiv(DIM * DIM, 256), 256, 0, stream>>>(Wk, wtk, DIM, DIM);
    transpose_w_kernel<<<cdiv(DIM * DIM, 256), 256, 0, stream>>>(Wv, wtv, DIM, DIM);
    transpose_w_kernel<<<cdiv(DIM * DIM, 256), 256, 0, stream>>>(Wpe2, wtpe2, DIM, DIM);
    transpose_w_kernel<<<cdiv(DIM * DIM, 256), 256, 0, stream>>>(Wg, wtg, DIM, DIM);
    transpose_w_kernel<<<cdiv(DIM * DIM, 256), 256, 0, stream>>>(Wo, wto, DIM, DIM);
    transpose_w_kernel<<<cdiv(DIM * D4, 256), 256, 0, stream>>>(Wf1, wtf1, DIM, D4);
    transpose_w_kernel<<<cdiv(D4 * D2, 256), 256, 0, stream>>>(Wf2, wtf2, D4, D2);
    transpose_wpool_kernel<<<cdiv(D2 * PKK, 256), 256, 0, stream>>>(Wpool, wtpool);

    // 2) Q/K/V projections (WMMA, B tile via TDM->LDS)
    wmma_gemm_kernel<<<dim3(BN / 16, DIM / 64), B128, blds(DIM), stream>>>(
        xt16, wtq, bq, Qf, (half_t*)nullptr, (const float*)nullptr, BN, DIM, DIM, 0);
    wmma_gemm_kernel<<<dim3(BN / 16, DIM / 64), B128, blds(DIM), stream>>>(
        xt16, wtk, bk, Kf, (half_t*)nullptr, (const float*)nullptr, BN, DIM, DIM, 0);
    wmma_gemm_kernel<<<dim3(BN / 16, DIM / 64), B128, blds(DIM), stream>>>(
        xt16, wtv, bv, Vf, (half_t*)nullptr, (const float*)nullptr, BN, DIM, DIM, 0);

    // 3) KNN of points against themselves (async LDS staging)
    knn_kernel<<<dim3(NPTS / 256, NB), 256, 0, stream>>>(p_pos, p_pos, idx1, NPTS);

    // 4) pe hidden (scalar 3->128 matvec + gelu, f16), then pe = peh @ Wpe2 (WMMA)
    pe_hidden_kernel<<<ATTROWS, DIM, 0, stream>>>(p_pos, idx1, Wpe1, bpe1, peh16);
    wmma_gemm_kernel<<<dim3(ATTROWS / 16, DIM / 64), B128, blds(DIM), stream>>>(
        peh16, wtpe2, bpe2, pe_f, (half_t*)nullptr, (const float*)nullptr, ATTROWS, DIM, DIM, 0);

    // 5) attn logits: (q - k_knn + pe) @ Wg (WMMA), softmax over feature dim
    attn_in_kernel<<<ATTROWS, DIM, 0, stream>>>(Qf, Kf, pe_f, idx1, ain16);
    wmma_gemm_kernel<<<dim3(ATTROWS / 16, DIM / 64), B128, blds(DIM), stream>>>(
        ain16, wtg, bg, attn_f, (half_t*)nullptr, (const float*)nullptr, ATTROWS, DIM, DIM, 0);
    softmax_kernel<<<ATTROWS, DIM, 0, stream>>>(attn_f);

    // 6) weighted sum over neighbors, then x1 = x + res @ Wo (WMMA + residual)
    res_kernel<<<BN, DIM, 0, stream>>>(attn_f, Vf, pe_f, idx1, res16);
    wmma_gemm_kernel<<<dim3(BN / 16, DIM / 64), B128, blds(DIM), stream>>>(
        res16, wto, bo, x1, (half_t*)nullptr, x, BN, DIM, DIM, 0);

    // 7) RMSNorm -> FFN (gelu on first GEMM, f16 intermediate)
    rmsnorm_kernel<<<BN, DIM, 0, stream>>>(x1, scale, xn16);
    wmma_gemm_kernel<<<dim3(BN / 16, D4 / 64), B128, blds(DIM), stream>>>(
        xn16, wtf1, bf1, (float*)nullptr, f116, (const float*)nullptr, BN, D4, DIM, 1);
    wmma_gemm_kernel<<<dim3(BN / 16, D2 / 64), B128, blds(D4), stream>>>(
        f116, wtf2, bf2, x2f, (half_t*)nullptr, (const float*)nullptr, BN, D2, D4, 0);

    // 8) pool: FPS, KNN to sampled points, pack, GEMM with leaky ReLU, max
    fps_kernel<<<NB, 1024, 0, stream>>>(p_pos, out_fp);
    knn_kernel<<<dim3(MOUT / 256, NB), 256, 0, stream>>>(out_fp, p_pos, kidx2, MOUT);
    pool_pack_kernel<<<cdiv((size_t)POOLROWS * PKK, 256), 256, 0, stream>>>(
        out_fp, x2f, kidx2, apool);
    wmma_gemm_kernel<<<dim3(POOLROWS / 16, D2 / 64), B128, blds(PKK), stream>>>(
        apool, wtpool, (const float*)nullptr, hpool, (half_t*)nullptr,
        (const float*)nullptr, POOLROWS, D2, PKK, 2);
    pool_max_kernel<<<cdiv(NB * MOUT * D2, 256), 256, 0, stream>>>(hpool, out_h);
}
